// ChildSumTreeLSTMEncoder_69020124447164
// MI455X (gfx1250) — compile-verified, matrix operand order checked by reading, not run on hardware
//
#include <hip/hip_runtime.h>
#include <hip/hip_bf16.h>

// ---------------- Types ----------------
typedef __attribute__((ext_vector_type(16))) __bf16          v16bf;
typedef __attribute__((ext_vector_type(16))) unsigned short  v16u;
typedef __attribute__((ext_vector_type(8)))  float           v8f;
typedef __attribute__((ext_vector_type(8)))  unsigned int    v8ui;

#define NTOT 87381L   // total nodes per batch element
#define EDIM 128
#define HDIM 128
#define FOURH 512

// ---------------- Helpers ----------------
__device__ __forceinline__ unsigned short f2bf(float f) {
  unsigned int u = __float_as_uint(f);
  u += 0x7FFFu + ((u >> 16) & 1u);          // round-to-nearest-even
  return (unsigned short)(u >> 16);
}
__device__ __forceinline__ float bf2f(unsigned short h) {
  return __uint_as_float(((unsigned int)h) << 16);
}
__device__ __forceinline__ float sigm(float x) { return 1.0f / (1.0f + __expf(-x)); }

// Build a 16x32 bf16 A fragment from LDS (row-major, `stride` elements/row).
// Layout per CDNA5 ISA 7.12.2: lane m(0-15)=row m, K 0..7,16..23; lane m+16: K 8..15,24..31.
__device__ __forceinline__ v16bf load_a_frag(const unsigned short* s, int row0,
                                             int stride, int kb, int lane) {
  const int m = lane & 15, khalf = lane >> 4;
  const unsigned short* rp = s + (row0 + m) * stride + kb * 32 + khalf * 8;
  v16u r;
#pragma unroll
  for (int e = 0; e < 16; ++e) r[e] = rp[((e < 8) ? 0 : 16) + (e & 7)];
  return __builtin_bit_cast(v16bf, r);
}

// Load a pre-packed 32x16 bf16 B fragment from global (1KB per fragment, lane-major).
__device__ __forceinline__ v16bf load_b_frag(const unsigned short* wp, int ct,
                                             int kb, int numKb, int lane) {
  const unsigned short* p = wp + ((size_t)(ct * numKb + kb) << 9) + (lane << 4);
  v8ui q = *(const v8ui*)p;                  // 32B vector load -> 2x b128
  return __builtin_bit_cast(v16bf, q);
}

__device__ __forceinline__ v8f wmma_bf16(v16bf a, v16bf b, v8f c) {
  return __builtin_amdgcn_wmma_f32_16x16x32_bf16(false, a, false, b, (short)0, c,
                                                 false, false);
}

// ---------------- Weight pack: f32 (KxN row-major) -> bf16 B-fragments ----------------
__global__ void pack_weights_kernel(const float* __restrict__ W,
                                    unsigned short* __restrict__ dst,
                                    int K, int Ncols) {
  const int numKb = K >> 5;
  const int total = K * Ncols;
  for (int d = blockIdx.x * blockDim.x + threadIdx.x; d < total;
       d += gridDim.x * blockDim.x) {
    int e = d & 15;
    int lane = (d >> 4) & 31;
    int frag = d >> 9;
    int kb = frag % numKb;
    int ct = frag / numKb;
    int col = ct * 16 + (lane & 15);
    int khalf = lane >> 4;
    int k = kb * 32 + ((e < 8) ? 0 : 16) + khalf * 8 + (e & 7);
    dst[d] = f2bf(W[(size_t)k * Ncols + col]);
  }
}

// ---------------- Fused per-level TreeLSTM kernel ----------------
// Block = 256 threads (8 waves), owns 16 flattened (batch,node) rows.
__global__ __launch_bounds__(256) void tree_lstm_level_kernel(
    const float* __restrict__ x,          // (B, NTOT, E) f32
    const unsigned short* __restrict__ wxp,
    const unsigned short* __restrict__ uioup,
    const unsigned short* __restrict__ ufp,
    const float* __restrict__ bias,       // (512,)
    const unsigned short* __restrict__ h_prev,  // bf16, (B*4*n_l, H)
    const float* __restrict__ c_prev,           // f32,  (B*4*n_l, H)
    unsigned short* __restrict__ h_cur,         // bf16, (B*n_l, H)
    float* __restrict__ c_cur,                  // f32,  (B*n_l, H)
    float* __restrict__ out,                    // non-null only at root
    int n_l, int shift, long node_off, int level) {
  __shared__ __align__(16) unsigned char smem[62720];
  float*          s_pre  = (float*)smem;                       // 16x512 f32  (32768B)
  unsigned short* s_hc   = (unsigned short*)(smem + 32768);    // 64x136 bf16 (17408B)
  unsigned short* s_hsum = (unsigned short*)(smem + 50176);    // 16x136 bf16 ( 4352B)
  unsigned short* s_x    = (unsigned short*)(smem + 54528);    // 16x136 bf16 (overlaps fsum)
  float*          s_fsum = (float*)(smem + 54528);             // 16x128 f32  ( 8192B)

  const int tid = threadIdx.x;
  const int lane = tid & 31;
  const int w = tid >> 5;
  const int M = 4 * n_l;              // B * n_l flattened rows
  const int r0 = blockIdx.x * 16;
  const bool leaf = (level == 0);
  const long nprev = 4L * n_l;

  // ---- Stage x tile (f32 -> bf16) ----
#pragma unroll
  for (int q = 0; q < 8; ++q) {
    int idx = q * 256 + tid;          // 2048 = 16*128
    int row = idx >> 7, col = idx & 127;
    int r = r0 + row;
    float v = 0.f;
    if (r < M) {
      int b = r >> shift;
      int node = r & (n_l - 1);
      v = x[((long)b * NTOT + node_off + node) * EDIM + col];
    }
    s_x[row * 136 + col] = f2bf(v);
  }
  // ---- Stage children h tile (bf16 copy), 64x128 ----
  if (!leaf) {
#pragma unroll
    for (int q = 0; q < 32; ++q) {
      int idx = q * 256 + tid;        // 8192 = 64*128
      int crow = idx >> 7, col = idx & 127;
      int t = crow >> 2, ch = crow & 3;
      int r = r0 + t;
      unsigned short v = 0;
      if (r < M) {
        int b = r >> shift;
        int node = r & (n_l - 1);
        v = h_prev[((long)b * nprev + (long)node * 4 + ch) * HDIM + col];
      }
      s_hc[crow * 136 + col] = v;
    }
  }
  __syncthreads();

  // ---- x GEMM: pre = x@Wx + b  (each wave: 4 column tiles, K-loop of 4 WMMAs) ----
#pragma unroll
  for (int i = 0; i < 4; ++i) {
    const int ct = w + i * 8;
    v8f acc = {};
#pragma unroll
    for (int kb = 0; kb < 4; ++kb) {
      v16bf a = load_a_frag(s_x, 0, 136, kb, lane);
      v16bf b = load_b_frag(wxp, ct, kb, 4, lane);
      acc = wmma_bf16(a, b, acc);
    }
    const int coln = ct * 16 + (lane & 15);
    const float bv = bias[coln];
    const int mbase = (lane >= 16) ? 8 : 0;
#pragma unroll
    for (int v = 0; v < 8; ++v) s_pre[(mbase + v) * FOURH + coln] = acc[v] + bv;
  }
  __syncthreads();

  if (!leaf) {
    // ---- hsum (bf16) + zero fsum ----
#pragma unroll
    for (int q = 0; q < 8; ++q) {
      int idx = q * 256 + tid;
      int row = idx >> 7, col = idx & 127;
      float s = bf2f(s_hc[(row * 4 + 0) * 136 + col]) +
                bf2f(s_hc[(row * 4 + 1) * 136 + col]) +
                bf2f(s_hc[(row * 4 + 2) * 136 + col]) +
                bf2f(s_hc[(row * 4 + 3) * 136 + col]);
      s_hsum[row * 136 + col] = f2bf(s);
      s_fsum[idx] = 0.f;
    }
    __syncthreads();

    // ---- Uiou GEMM: hsum @ Uiou, accumulate into i/o/u slices of s_pre ----
#pragma unroll
    for (int i = 0; i < 3; ++i) {
      const int ct = w + i * 8;       // 0..23 over 384 cols
      v8f acc = {};
#pragma unroll
      for (int kb = 0; kb < 4; ++kb) {
        v16bf a = load_a_frag(s_hsum, 0, 136, kb, lane);
        v16bf b = load_b_frag(uioup, ct, kb, 4, lane);
        acc = wmma_bf16(a, b, acc);
      }
      const int gate = ct >> 3, within = ct & 7;
      // gate order in pre is [i,f,o,u]; Uiou produces [i,o,u]
      const int pct = (gate == 0) ? within : (gate == 1) ? (16 + within) : (24 + within);
      const int coln = pct * 16 + (lane & 15);
      const int mbase = (lane >= 16) ? 8 : 0;
#pragma unroll
      for (int v = 0; v < 8; ++v) s_pre[(mbase + v) * FOURH + coln] += acc[v];
    }

    // ---- Uf GEMM (64 child rows x 128), fused f = sig(xf + .)*c_child reduction ----
    // wave w owns column tile w -> fsum columns [w*16, w*16+16): race-free.
#pragma unroll
    for (int rb = 0; rb < 4; ++rb) {
      v8f acc = {};
#pragma unroll
      for (int kb = 0; kb < 4; ++kb) {
        v16bf a = load_a_frag(s_hc, rb * 16, 136, kb, lane);
        v16bf b = load_b_frag(ufp, w, kb, 4, lane);
        acc = wmma_bf16(a, b, acc);
      }
      const int j = w * 16 + (lane & 15);
      const int mbase = (lane >= 16) ? 8 : 0;
#pragma unroll
      for (int v = 0; v < 8; ++v) {
        int crow = rb * 16 + mbase + v;
        int t = crow >> 2, ch = crow & 3;
        int r = r0 + t;
        if (r < M) {
          int b_ = r >> shift;
          int node = r & (n_l - 1);
          float ccv = c_prev[((long)b_ * nprev + (long)node * 4 + ch) * HDIM + j];
          float f = sigm(s_pre[t * FOURH + 128 + j] + acc[v]);
          s_fsum[t * HDIM + j] += f * ccv;
        }
      }
    }
  }
  __syncthreads();

  // ---- Fused activation epilogue ----
#pragma unroll
  for (int q = 0; q < 8; ++q) {
    int idx = q * 256 + tid;
    int t = idx >> 7, j = idx & 127;
    int r = r0 + t;
    if (r >= M) continue;
    float pi = s_pre[t * FOURH + j];
    float po = s_pre[t * FOURH + 256 + j];
    float pu = s_pre[t * FOURH + 384 + j];
    float ig = sigm(pi), og = sigm(po), ug = tanhf(pu);
    float c = ig * ug + (leaf ? 0.f : s_fsum[t * HDIM + j]);
    float h = og * tanhf(c);
    c_cur[(long)r * HDIM + j] = c;
    h_cur[(long)r * HDIM + j] = f2bf(h);
    if (out) {                        // root: n_l==1 -> r == batch index
      out[r * HDIM + j] = h;
      out[512 + r * HDIM + j] = c;
    }
  }
}

// ---------------- Host launch ----------------
extern "C" void kernel_launch(void* const* d_in, const int* in_sizes, int n_in,
                              void* d_out, int out_size, void* d_ws, size_t ws_size,
                              hipStream_t stream) {
  (void)in_sizes; (void)n_in; (void)out_size; (void)ws_size;
  const float* x    = (const float*)d_in[0];
  const float* Wx   = (const float*)d_in[1];
  const float* Uiou = (const float*)d_in[2];
  const float* Uf   = (const float*)d_in[3];
  const float* bias = (const float*)d_in[4];
  float* out = (float*)d_out;

  unsigned char* ws = (unsigned char*)d_ws;
  size_t off = 0;
  auto carve = [&](size_t bytes) -> void* {
    void* p = ws + off;
    off += (bytes + 255) & ~(size_t)255;
    return p;
  };
  unsigned short* wxp   = (unsigned short*)carve((size_t)128 * 512 * 2);
  unsigned short* uioup = (unsigned short*)carve((size_t)128 * 384 * 2);
  unsigned short* ufp   = (unsigned short*)carve((size_t)128 * 128 * 2);
  unsigned short* h0 = (unsigned short*)carve((size_t)4 * 65536 * 128 * 2);
  unsigned short* h1 = (unsigned short*)carve((size_t)4 * 16384 * 128 * 2);
  float* c0 = (float*)carve((size_t)4 * 65536 * 128 * 4);
  float* c1 = (float*)carve((size_t)4 * 16384 * 128 * 4);

  pack_weights_kernel<<<128, 256, 0, stream>>>(Wx, wxp, 128, 512);
  pack_weights_kernel<<<96, 256, 0, stream>>>(Uiou, uioup, 128, 384);
  pack_weights_kernel<<<32, 256, 0, stream>>>(Uf, ufp, 128, 128);

  long offsets[10];
  offsets[0] = 0;
  for (int l = 0; l <= 8; ++l) offsets[l + 1] = offsets[l] + (1L << (2 * (8 - l)));

  for (int l = 0; l <= 8; ++l) {
    int nl = 1 << (2 * (8 - l));
    int M = 4 * nl;
    int blocks = (M + 15) / 16;
    unsigned short* hc = (l & 1) ? h1 : h0;
    float*          cc = (l & 1) ? c1 : c0;
    unsigned short* hp = (l & 1) ? h0 : h1;
    float*          cp = (l & 1) ? c0 : c1;
    tree_lstm_level_kernel<<<blocks, 256, 0, stream>>>(
        x, wxp, uioup, ufp, bias,
        (l == 0) ? nullptr : hp, (l == 0) ? nullptr : cp,
        hc, cc, (l == 8) ? out : nullptr,
        nl, 2 * (8 - l), offsets[l], l);
  }
}
